// GS_TCM_49881750176390
// MI455X (gfx1250) — matrix operand order repaired and checked by reading, not compiled, not run on hardware
//
#include <hip/hip_runtime.h>

// ---------------------------------------------------------------- types
typedef __attribute__((ext_vector_type(16))) __bf16 v16bf;
typedef __attribute__((ext_vector_type(8)))  __bf16 v8bf;
typedef __attribute__((ext_vector_type(8)))  float  v8f;

#define CEILDIV(a,b) (((a)+(b)-1)/(b))

__device__ inline __bf16 f2bf(float f) {
  unsigned u = __float_as_uint(f);
  u += 0x7FFFu + ((u >> 16) & 1u);            // round-to-nearest-even
  unsigned short h = (unsigned short)(u >> 16);
  return __builtin_bit_cast(__bf16, h);
}

// ---------------------------------------------------------------- generic batched WMMA GEMM
// C[M,N] = (accum ? C : 0) + A[M,K] * B[K,N], f32 in/out, bf16 WMMA compute.
// transA: A(m,k) = A[k*lda+m] else A[m*lda+k];  transB: B(k,n) = B[n*ldb+k] else B[k*ldb+n]
// blockIdx.z decomposed: z1 = z % nz1, z0 = z / nz1; per-matrix pointer offsets (floats).
// One wave per 16x32 C macro-tile (two accumulators sharing the A fragment);
// K staged through LDS in 64-wide macro-steps -> 4 v_wmma_f32_16x16x32_bf16 per stage,
// fragments read as ds_load_b128. Transpose handled by hoisted strides (branchless fill);
// out-of-range M/N use clamped addresses (unstored rows/cols), K padding selected to 0.
__global__ __launch_bounds__(32)
void k_gemm_wmma(const float* __restrict__ A, const float* __restrict__ B,
                 float* __restrict__ C,
                 int M, int N, int K, int lda, int ldb, int ldc,
                 int transA, int transB, int accum, int nz1,
                 long sA0, long sA1, long sB0, long sB1, long sC0, long sC1)
{
  int z  = blockIdx.z;
  int z1 = z % nz1, z0 = z / nz1;
  A += (size_t)z0 * sA0 + (size_t)z1 * sA1;
  B += (size_t)z0 * sB0 + (size_t)z1 * sB1;
  C += (size_t)z0 * sC0 + (size_t)z1 * sC1;

  const int m0   = blockIdx.x * 16;
  const int n0   = blockIdx.y * 32;
  const int lane = threadIdx.x;
  const int half = lane >> 4;          // 0/1
  const int l16  = lane & 15;

  // hoisted transpose strides: A(m,k) = A[m*amS + k*akS], B(k,n) = B[k*bkS + n*bnS]
  const size_t amS = transA ? 1 : (size_t)lda;
  const size_t akS = transA ? (size_t)lda : 1;
  const size_t bkS = transB ? 1 : (size_t)ldb;
  const size_t bnS = transB ? (size_t)ldb : 1;

  __shared__ __bf16 As[16][64];        // [m][k]  row = 128B, 16B chunks aligned
  __shared__ __bf16 Bs[32][64];        // [n][k]  transposed for contiguous frag reads

  v8f acc0 = {}, acc1 = {};

  for (int kb = 0; kb < K; kb += 64) {
    // ---- A tile fill: lane = k column -> coalesced for row-major A
    for (int r = 0; r < 16; ++r) {
#pragma unroll
      for (int j = 0; j < 2; ++j) {
        int mr = m0 + r, kr = kb + j * 32 + lane;
        int mc = mr < M ? mr : M - 1;
        int kc = kr < K ? kr : K - 1;
        float v = A[(size_t)mc * amS + (size_t)kc * akS];
        if (kr >= K) v = 0.f;          // K padding must be zero; M clamp is harmless
        As[r][j * 32 + lane] = f2bf(v);
      }
    }
    // ---- B tile fill (stored transposed): lane = n column -> coalesced for row-major B
#pragma unroll
    for (int g = 0; g < 2; ++g) {
      for (int j = 0; j < 32; ++j) {
        int kr = kb + j * 2 + half;
        int nr = n0 + g * 16 + l16;
        int kc = kr < K ? kr : K - 1;
        int nc = nr < N ? nr : N - 1;
        float v = B[(size_t)kc * bkS + (size_t)nc * bnS];
        if (kr >= K) v = 0.f;
        Bs[g * 16 + l16][j * 2 + half] = f2bf(v);
      }
    }
    if (kb + 64 < K) {                 // gfx1250 global_prefetch of next A chunk
      int mc = (m0 + l16) < M ? (m0 + l16) : M - 1;
      __builtin_prefetch(&A[(size_t)mc * amS + (size_t)(kb + 64) * akS], 0, 1);
    }
    __syncthreads();

    // ---- fragments in contiguous k order, same for A and B (wave32 16-bit A 16x32 / B 32x16):
    // A: lane holds row M=l16, K = 32s + half*16 + {0..15}            -> two b128 reads
    // B: lane holds col N=l16, K = 32s + half*16 + {0..15}            -> two b128 reads
    const v8bf* pa  = (const v8bf*)&As[l16][0];
    const v8bf* pb0 = (const v8bf*)&Bs[l16][0];
    const v8bf* pb1 = (const v8bf*)&Bs[16 + l16][0];
#pragma unroll
    for (int s = 0; s < 2; ++s) {
      v16bf a  = __builtin_shufflevector(pa[s * 4 + half * 2], pa[s * 4 + half * 2 + 1],
                                         0,1,2,3,4,5,6,7,8,9,10,11,12,13,14,15);
      v16bf b0 = __builtin_shufflevector(pb0[s * 4 + half * 2], pb0[s * 4 + half * 2 + 1],
                                         0,1,2,3,4,5,6,7,8,9,10,11,12,13,14,15);
      v16bf b1 = __builtin_shufflevector(pb1[s * 4 + half * 2], pb1[s * 4 + half * 2 + 1],
                                         0,1,2,3,4,5,6,7,8,9,10,11,12,13,14,15);
      acc0 = __builtin_amdgcn_wmma_f32_16x16x32_bf16(false, a, false, b0,
                                                     (short)0, acc0, false, false);
      acc1 = __builtin_amdgcn_wmma_f32_16x16x32_bf16(false, a, false, b1,
                                                     (short)0, acc1, false, false);
    }
    __syncthreads();
  }

#pragma unroll
  for (int r = 0; r < 8; ++r) {
    int m = m0 + half * 8 + r;         // C/D layout: VGPR r -> M = r + 8*half
    if (m < M) {
      int na = n0 + l16, nb = n0 + 16 + l16;
      if (na < N) {
        size_t ia = (size_t)m * ldc + na;
        C[ia] = accum ? (C[ia] + acc0[r]) : acc0[r];
      }
      if (nb < N) {
        size_t ib = (size_t)m * ldc + nb;
        C[ib] = accum ? (C[ib] + acc1[r]) : acc1[r];
      }
    }
  }
}

// ---------------------------------------------------------------- small kernels
__global__ void k_supports_init(const float* __restrict__ adj, float* __restrict__ S) {
  long i = blockIdx.x * (long)blockDim.x + threadIdx.x;
  if (i >= 250000) return;
  int m = i % 500, n = i / 500;
  S[i]          = (n == m) ? 1.f : 0.f;   // S0 = I
  S[250000 + i] = adj[i];                 // S1 = adj
}

__global__ void k_s2_fix(float* __restrict__ S2) {
  long i = blockIdx.x * (long)blockDim.x + threadIdx.x;
  if (i >= 250000) return;
  int m = i % 500, n = i / 500;
  S2[i] = 2.f * S2[i] - ((n == m) ? 1.f : 0.f);
}

__global__ void k_zero(float* p, long n) {
  long i = blockIdx.x * (long)blockDim.x + threadIdx.x;
  if (i < n) p[i] = 0.f;
}

__global__ void k_pool_mean(const float* __restrict__ x, float* __restrict__ pooled,
                            int C, long sBT, long sC, long sN) {
  int i = blockIdx.x * blockDim.x + threadIdx.x;
  if (i >= 192 * C) return;
  int c = i % C, bt = i / C;
  const float* p = x + (size_t)bt * sBT + (size_t)c * sC;
  float s = 0.f;
  for (int n = 0; n < 500; ++n) s += p[(size_t)n * sN];
  pooled[i] = s * (1.f / 500.f);
}

__global__ void k_mix(const float* __restrict__ pooled,
                      const float* __restrict__ f1w, const float* __restrict__ f1b,
                      const float* __restrict__ f2w, const float* __restrict__ f2b,
                      float* __restrict__ a, int C) {
  int bt = blockIdx.x * blockDim.x + threadIdx.x;
  if (bt >= 192) return;
  float hid[8];
  for (int k = 0; k < 8; ++k) {
    float s = f1b[k];
    for (int c = 0; c < C; ++c) s += pooled[bt * C + c] * f1w[k * C + c];
    hid[k] = fmaxf(s, 0.f);
  }
  float lg[8], mx = -1e30f;
  for (int j = 0; j < 8; ++j) {
    float s = f2b[j];
    for (int k = 0; k < 8; ++k) s += hid[k] * f2w[j * 8 + k];
    lg[j] = s; mx = fmaxf(mx, s);
  }
  float den = 0.f;
  for (int j = 0; j < 8; ++j) { lg[j] = __expf(lg[j] - mx); den += lg[j]; }
  float inv = 1.f / den;
  for (int j = 0; j < 8; ++j) a[bt * 8 + j] = lg[j] * inv;
}

__global__ void k_ne(const float* __restrict__ a, const float* __restrict__ emb,
                     float* __restrict__ ne) {
  long i = blockIdx.x * (long)blockDim.x + threadIdx.x;
  if (i >= 192L * 500 * 8) return;
  int d = i % 8; long r = i / 8; int n = r % 500; int bt = r / 500;
  float s = 0.f;
  for (int k = 0; k < 8; ++k) s += a[bt * 8 + k] * emb[((size_t)k * 500 + n) * 8 + d];
  ne[i] = s;
}

// wpk[k][c][d*128+o] = wp[d][k][c][o]
__global__ void k_repack_wp(const float* __restrict__ wp, float* __restrict__ wpk, int C) {
  long i = blockIdx.x * (long)blockDim.x + threadIdx.x;
  long tot = 3L * C * 8 * 128;
  if (i >= tot) return;
  int o = i % 128; long r = i / 128;
  int d = r % 8; r /= 8; int c = r % C; int k = r / C;
  wpk[((size_t)(k * C + c)) * 1024 + d * 128 + o] =
      wp[(((size_t)d * 3 + k) * C + c) * 128 + o];
}

// y[bt][n][o] = sum_d ne[bt][n][d]*(tmp[btL][n][d][o] + bp[d][o])
__global__ void k_combine(const float* __restrict__ tmp, const float* __restrict__ ne,
                          const float* __restrict__ bp, float* __restrict__ y,
                          int bt0, int G) {
  long i = blockIdx.x * (long)blockDim.x + threadIdx.x;
  long tot = (long)G * 500 * 128;
  if (i >= tot) return;
  int o = i % 128; long r = i / 128; int n = r % 500; int btL = r / 500;
  int bt = bt0 + btL;
  const float* tp = tmp + ((size_t)(btL * 500 + n)) * 1024 + o;
  const float* np = ne + ((size_t)((size_t)bt * 500 + n)) * 8;
  float s = 0.f;
#pragma unroll
  for (int d = 0; d < 8; ++d) s += np[d] * (tp[d * 128] + bp[d * 128 + o]);
  y[((size_t)((size_t)bt * 500 + n)) * 128 + o] = s;
}

__global__ void k_bn_stats(const float* __restrict__ x, int count, int C,
                           float* __restrict__ mean, float* __restrict__ rstd, float eps) {
  int c = blockIdx.x;
  float s = 0.f, s2 = 0.f;
  for (int i = threadIdx.x; i < count; i += blockDim.x) {
    float v = x[(size_t)i * C + c];
    s += v; s2 += v * v;
  }
  __shared__ float sh[256], sh2[256];
  sh[threadIdx.x] = s; sh2[threadIdx.x] = s2;
  __syncthreads();
  for (int o = blockDim.x / 2; o > 0; o >>= 1) {
    if ((int)threadIdx.x < o) { sh[threadIdx.x] += sh[threadIdx.x + o];
                                sh2[threadIdx.x] += sh2[threadIdx.x + o]; }
    __syncthreads();
  }
  if (threadIdx.x == 0) {
    float m = sh[0] / count;
    float v = sh2[0] / count - m * m;
    mean[c] = m; rstd[c] = rsqrtf(v + eps);
  }
}

__global__ void k_bn_apply(float* __restrict__ x, const float* __restrict__ mean,
                           const float* __restrict__ rstd, const float* __restrict__ g,
                           const float* __restrict__ b, long total, int C, int relu) {
  long i = blockIdx.x * (long)blockDim.x + threadIdx.x;
  if (i >= total) return;
  int c = i % C;
  float v = (x[i] - mean[c]) * rstd[c] * g[c] + b[c];
  if (relu) v = fmaxf(v, 0.f);
  x[i] = v;
}

__global__ void k_add_bias_add(float* __restrict__ dst, const float* __restrict__ bias,
                               const float* __restrict__ other, long total, int C) {
  long i = blockIdx.x * (long)blockDim.x + threadIdx.x;
  if (i >= total) return;
  dst[i] = dst[i] + bias[i % C] + other[i];
}

__global__ void k_gru_step(const float* __restrict__ gi, const float* __restrict__ gh,
                           const float* __restrict__ bih, const float* __restrict__ bhh,
                           float* __restrict__ h, float* __restrict__ gout, int t) {
  long i = blockIdx.x * (long)blockDim.x + threadIdx.x;
  if (i >= 4000L * 128) return;
  int hh = i % 128; int s = i / 128;
  float ir = gi[(size_t)s * 384 + hh]        + bih[hh];
  float iz = gi[(size_t)s * 384 + 128 + hh]  + bih[128 + hh];
  float in_= gi[(size_t)s * 384 + 256 + hh]  + bih[256 + hh];
  float hr = gh[(size_t)s * 384 + hh]        + bhh[hh];
  float hz = gh[(size_t)s * 384 + 128 + hh]  + bhh[128 + hh];
  float hn = gh[(size_t)s * 384 + 256 + hh]  + bhh[256 + hh];
  float r = 1.f / (1.f + __expf(-(ir + hr)));
  float z = 1.f / (1.f + __expf(-(iz + hz)));
  float n = tanhf(in_ + r * hn);
  float hnew = (1.f - z) * n + z * h[i];
  h[i] = hnew;
  gout[((size_t)s * 24 + t) * 128 + hh] = hnew;
}

// temporal conv, kernel 3, pad 1 over t.  x,y: [b][n][t][128]; w: [ho][c][3]
__global__ void k_conv_t(const float* __restrict__ x, const float* __restrict__ w,
                         const float* __restrict__ bias, float* __restrict__ y) {
  long i = blockIdx.x * (long)blockDim.x + threadIdx.x;
  if (i >= 96000L * 128) return;
  int ho = i % 128; long r = i / 128; int t = r % 24; long bn = r / 24;
  const float* xp = x + (size_t)bn * 24 * 128;
  float s = bias[ho];
  for (int dt = 0; dt < 3; ++dt) {
    int tt = t + dt - 1;
    if (tt < 0 || tt >= 24) continue;
    const float* xr = xp + (size_t)tt * 128;
    const float* wr = w + (size_t)ho * 128 * 3 + dt;
    for (int c = 0; c < 128; ++c) s += xr[c] * wr[c * 3];
  }
  y[i] = s;
}

// per (b,n,head) attention; block = 1 wave
__global__ __launch_bounds__(32)
void k_attn(const float* __restrict__ q, const float* __restrict__ k,
            const float* __restrict__ v, const float* __restrict__ vb,
            float* __restrict__ xa) {
  int g = blockIdx.x;               // (b*500+n)*8 + h
  int h = g & 7; int bn = g >> 3;
  __shared__ float Q[24][16], K[24][16], V[24][16], P[24][24];
  int lane = threadIdx.x;
  const size_t base = (size_t)bn * 24 * 128 + h * 16;
  for (int j = lane; j < 384; j += 32) {
    int t = j >> 4, d = j & 15;
    size_t idx = base + (size_t)t * 128 + d;
    Q[t][d] = q[idx]; K[t][d] = k[idx]; V[t][d] = v[idx] + vb[h * 16 + d];
  }
  __syncthreads();
  for (int j = lane; j < 576; j += 32) {
    int i = j / 24, jj = j % 24;
    float s = 0.f;
#pragma unroll
    for (int d = 0; d < 16; ++d) s += Q[i][d] * K[jj][d];
    P[i][jj] = s * 0.25f;           // / sqrt(dk=16)
  }
  __syncthreads();
  if (lane < 24) {
    float mx = -1e30f;
    for (int jj = 0; jj < 24; ++jj) mx = fmaxf(mx, P[lane][jj]);
    float den = 0.f;
    for (int jj = 0; jj < 24; ++jj) { float e = __expf(P[lane][jj] - mx); P[lane][jj] = e; den += e; }
    float inv = 1.f / den;
    for (int jj = 0; jj < 24; ++jj) P[lane][jj] *= inv;
  }
  __syncthreads();
  for (int j = lane; j < 384; j += 32) {
    int t = j >> 4, d = j & 15;
    float s = 0.f;
    for (int jj = 0; jj < 24; ++jj) s += P[t][jj] * V[jj][d];
    xa[base + (size_t)t * 128 + d] = s;
  }
}

__global__ void k_fcreduce(const float* __restrict__ xo, const float* __restrict__ ob,
                           const float* __restrict__ fcw, const float* __restrict__ fcb,
                           float* __restrict__ out2f) {
  long i = blockIdx.x * (long)blockDim.x + threadIdx.x;
  if (i >= 4000L * 128) return;
  int d = i % 128; int bn = i / 128;
  float s = fcb[0];
  for (int t = 0; t < 24; ++t)
    s += (xo[((size_t)bn * 24 + t) * 128 + d] + ob[d]) * fcw[t];
  out2f[i] = s;
}

__global__ void k_resreduce(const float* __restrict__ r2, const float* __restrict__ r2b,
                            const float* __restrict__ rcw, const float* __restrict__ rcb,
                            float* __restrict__ resred) {
  long i = blockIdx.x * (long)blockDim.x + threadIdx.x;
  if (i >= 4000L * 128) return;
  int hh = i % 128; int bn = i / 128; int n = bn % 500; int b = bn / 500;
  float s = rcb[0];
  for (int t = 0; t < 24; ++t)
    s += (r2[(((size_t)(b * 24 + t)) * 500 + n) * 128 + hh] + r2b[hh]) * rcw[t];
  resred[i] = s;
}

__global__ void k_te(const float* __restrict__ tt, const float* __restrict__ w1,
                     const float* __restrict__ b1, const float* __restrict__ w2,
                     const float* __restrict__ b2, float* __restrict__ te) {
  int i = blockIdx.x * blockDim.x + threadIdx.x;
  if (i >= 1024) return;
  int hh = i % 128; int b = i / 128;
  float s = b2[hh];
  for (int j = 0; j < 128; ++j) {
    float hid = b1[j];
    for (int f = 0; f < 32; ++f) hid += tt[b * 32 + f] * w1[j * 32 + f];
    hid = fmaxf(hid, 0.f);
    s += hid * w2[hh * 128 + j];
  }
  te[i] = s;
}

__global__ void k_prefin(const float* __restrict__ out2f, const float* __restrict__ resbn,
                         const float* __restrict__ te, float* __restrict__ pf) {
  long i = blockIdx.x * (long)blockDim.x + threadIdx.x;
  if (i >= 4000L * 128) return;
  int hh = i % 128; int bn = i / 128; int b = bn / 500;
  pf[i] = out2f[i] + resbn[i] + te[b * 128 + hh];
}

__global__ void k_bias(float* __restrict__ x, const float* __restrict__ b, long total, int C) {
  long i = blockIdx.x * (long)blockDim.x + threadIdx.x;
  if (i >= total) return;
  x[i] += b[i % C];
}

__global__ void k_out(const float* __restrict__ gout, float* __restrict__ out) {
  long i = blockIdx.x * (long)blockDim.x + threadIdx.x;
  if (i >= 8L * 64 * 500) return;
  int g = i % 500; int e = (i / 500) % 64; int b = i / (500 * 64);
  out[i] = gout[((size_t)b * 500 + g) * 64 + e];
}

// ---------------------------------------------------------------- host
static inline void gemm(hipStream_t st, const float* A, const float* B, float* C,
                        int M, int N, int K, int lda, int ldb, int ldc,
                        int tA, int tB, int acc,
                        int gz = 1, int nz1 = 1,
                        long sA0 = 0, long sA1 = 0, long sB0 = 0, long sB1 = 0,
                        long sC0 = 0, long sC1 = 0) {
  dim3 g(CEILDIV(M, 16), CEILDIV(N, 32), gz);   // 16x32 macro-tile per wave
  k_gemm_wmma<<<g, 32, 0, st>>>(A, B, C, M, N, K, lda, ldb, ldc, tA, tB, acc, nz1,
                                sA0, sA1, sB0, sB1, sC0, sC1);
}

extern "C" void kernel_launch(void* const* d_in, const int* in_sizes, int n_in,
                              void* d_out, int out_size, void* d_ws, size_t ws_size,
                              hipStream_t stream) {
  const float* adj   = (const float*)d_in[0];
  const float* x     = (const float*)d_in[1];
  const float* gmap  = (const float*)d_in[2];
  const float* ttime = (const float*)d_in[3];
  const float* nemb  = (const float*)d_in[4];
  const float* wp0   = (const float*)d_in[5];
  const float* bp0   = (const float*)d_in[6];
  const float* a0f1w = (const float*)d_in[7];
  const float* a0f1b = (const float*)d_in[8];
  const float* a0f2w = (const float*)d_in[9];
  const float* a0f2b = (const float*)d_in[10];
  const float* wp1   = (const float*)d_in[11];
  const float* bp1   = (const float*)d_in[12];
  const float* a1f1w = (const float*)d_in[13];
  const float* a1f1b = (const float*)d_in[14];
  const float* a1f2w = (const float*)d_in[15];
  const float* a1f2b = (const float*)d_in[16];
  const float* bn0g  = (const float*)d_in[17];
  const float* bn0b  = (const float*)d_in[18];
  const float* bn1g  = (const float*)d_in[19];
  const float* bn1b  = (const float*)d_in[20];
  const float* wih   = (const float*)d_in[21];
  const float* whh   = (const float*)d_in[22];
  const float* bih   = (const float*)d_in[23];
  const float* bhh   = (const float*)d_in[24];
  const float* bngg  = (const float*)d_in[25];
  const float* bngb  = (const float*)d_in[26];
  const float* cqw   = (const float*)d_in[27];
  const float* cqb   = (const float*)d_in[28];
  const float* ckw   = (const float*)d_in[29];
  const float* ckb   = (const float*)d_in[30];
  const float* vw    = (const float*)d_in[31];
  const float* vb    = (const float*)d_in[32];
  const float* ow    = (const float*)d_in[33];
  const float* ob    = (const float*)d_in[34];
  const float* r1w   = (const float*)d_in[35];
  const float* r1b   = (const float*)d_in[36];
  const float* fcw   = (const float*)d_in[37];
  const float* fcb   = (const float*)d_in[38];
  const float* rcw   = (const float*)d_in[39];
  const float* rcb   = (const float*)d_in[40];
  const float* r2w   = (const float*)d_in[41];
  const float* r2b   = (const float*)d_in[42];
  const float* bnrg  = (const float*)d_in[43];
  const float* bnrb  = (const float*)d_in[44];
  const float* t1w   = (const float*)d_in[45];
  const float* t1b   = (const float*)d_in[46];
  const float* t2w   = (const float*)d_in[47];
  const float* t2b   = (const float*)d_in[48];
  const float* finw  = (const float*)d_in[49];
  const float* finb  = (const float*)d_in[50];
  float* out = (float*)d_out;

  // -------- workspace layout (floats)
  float* ws = (float*)d_ws;
  size_t off = 0;
  auto alloc = [&](size_t n) { float* p = ws + off; off += n; return p; };
  float* S      = alloc(3 * 250000);     // supports
  float* wpk    = alloc(3 * 128 * 1024); // repacked weight pool (max layer1)
  float* pooled = alloc(192 * 128);
  float* amix   = alloc(192 * 8);
  float* ne     = alloc(192L * 500 * 8);
  float* meanb  = alloc(128);
  float* rstdb  = alloc(128);
  float* te     = alloc(8 * 128);
  float* out2f  = alloc(4000 * 128);
  float* resred = alloc(4000 * 128);
  float* finout = alloc(4000 * 64);
  float* gout   = alloc(4000 * 64);
  float* hbuf   = alloc(4000 * 128);
  float* gi     = alloc(4000 * 384);
  float* gh     = alloc(4000 * 384);
  float* prefin = alloc(4000 * 128);
  float* xg     = alloc(3L * 24 * 500 * 128);  // per-chunk propagated features
  float* Rtmp   = alloc(12288000);             // tmp (chunk) -> V
  float* R0     = alloc(12288000);             // y0/out0 -> Q -> xo
  float* R1     = alloc(12288000);             // y1/out1 -> K -> res2out
  float* out2   = alloc(12288000);             // residual stream (persistent)
  float* R2     = alloc(12288000);             // gru_out -> xa
  if (ws_size < off * sizeof(float)) return;   // not enough scratch

  const int TPB = 256;
  const long E12M = 12288000L;

  // -------- supports: S0=I, S1=adj, S2=2*adj@adj-I
  k_supports_init<<<CEILDIV(250000, TPB), TPB, 0, stream>>>(adj, S);
  gemm(stream, adj, adj, S + 500000, 500, 500, 500, 500, 500, 500, 0, 0, 0);
  k_s2_fix<<<CEILDIV(250000, TPB), TPB, 0, stream>>>(S + 500000);

  // ================= FlexGCN layer 0 (in: x [bt][64][500]) -> R0 [bt][n][128]
  k_pool_mean<<<CEILDIV(192 * 64, TPB), TPB, 0, stream>>>(x, pooled, 64, 64L * 500, 500, 1);
  k_mix<<<CEILDIV(192, 64), 64, 0, stream>>>(pooled, a0f1w, a0f1b, a0f2w, a0f2b, amix, 64);
  k_ne<<<CEILDIV(192L * 500 * 8, TPB), TPB, 0, stream>>>(amix, nemb, ne);
  k_repack_wp<<<CEILDIV(3L * 64 * 8 * 128, TPB), TPB, 0, stream>>>(wp0, wpk, 64);
  for (int c = 0; c < 8; ++c) {
    int bt0 = c * 24;
    // xg[k][btL][n][cin] = S_k @ xr_bt   (B read transposed from x[bt][cin][n])
    gemm(stream, S, x + (size_t)bt0 * 64 * 500, xg, 500, 64, 500, 500, 500, 64,
         0, 1, 0, 72, 24, 250000, 0, 0, 64L * 500, 24L * 500 * 64, 500L * 64);
    for (int k = 0; k < 3; ++k)   // tmp[(btL,n)][d*128+o] += xg_k @ W_k
      gemm(stream, xg + (size_t)k * 24 * 500 * 64, wpk + (size_t)k * 64 * 1024, Rtmp,
           12000, 1024, 64, 64, 1024, 1024, 0, 0, (k > 0) ? 1 : 0);
    k_combine<<<CEILDIV(24L * 500 * 128, TPB), TPB, 0, stream>>>(Rtmp, ne, bp0, R0, bt0, 24);
  }
  k_bn_stats<<<128, 256, 0, stream>>>(R0, 96000, 128, meanb, rstdb, 1e-5f);
  k_bn_apply<<<CEILDIV(E12M, TPB), TPB, 0, stream>>>(R0, meanb, rstdb, bn0g, bn0b, E12M, 128, 1);

  // ================= FlexGCN layer 1 (in: R0 [bt][n][128]) -> R1
  k_pool_mean<<<CEILDIV(192 * 128, TPB), TPB, 0, stream>>>(R0, pooled, 128, 500L * 128, 1, 128);
  k_mix<<<CEILDIV(192, 64), 64, 0, stream>>>(pooled, a1f1w, a1f1b, a1f2w, a1f2b, amix, 128);
  k_ne<<<CEILDIV(192L * 500 * 8, TPB), TPB, 0, stream>>>(amix, nemb, ne);
  k_repack_wp<<<CEILDIV(3L * 128 * 8 * 128, TPB), TPB, 0, stream>>>(wp1, wpk, 128);
  for (int c = 0; c < 8; ++c) {
    int bt0 = c * 24;
    gemm(stream, S, R0 + (size_t)bt0 * 500 * 128, xg, 500, 128, 500, 500, 128, 128,
         0, 0, 0, 72, 24, 250000, 0, 0, 500L * 128, 24L * 500 * 128, 500L * 128);
    for (int k = 0; k < 3; ++k)
      gemm(stream, xg + (size_t)k * 24 * 500 * 128, wpk + (size_t)k * 128 * 1024, Rtmp,
           12000, 1024, 128, 128, 1024, 1024, 0, 0, (k > 0) ? 1 : 0);
    k_combine<<<CEILDIV(24L * 500 * 128, TPB), TPB, 0, stream>>>(Rtmp, ne, bp1, R1, bt0, 24);
  }
  k_bn_stats<<<128, 256, 0, stream>>>(R1, 96000, 128, meanb, rstdb, 1e-5f);
  k_bn_apply<<<CEILDIV(E12M, TPB), TPB, 0, stream>>>(R1, meanb, rstdb, bn1g, bn1b, E12M, 128, 1);

  // ================= residual 1: out2 = xr @ r1w^T + r1b + out1
  gemm(stream, x, r1w, out2, 500, 128, 64, 500, 64, 128,
       1, 1, 0, 192, 1, 64L * 500, 0, 0, 0, 500L * 128, 0);
  k_add_bias_add<<<CEILDIV(E12M, TPB), TPB, 0, stream>>>(out2, r1b, R1, E12M, 128);

  // ================= GRU over time -> R2 = [b][n][t][128]
  k_zero<<<CEILDIV(512000L, TPB), TPB, 0, stream>>>(hbuf, 512000);
  for (int t = 0; t < 24; ++t) {
    gemm(stream, out2 + (size_t)t * 500 * 128, wih, gi, 500, 384, 128, 128, 128, 384,
         0, 1, 0, 8, 1, 24L * 500 * 128, 0, 0, 0, 500L * 384, 0);
    gemm(stream, hbuf, whh, gh, 4000, 384, 128, 128, 128, 384, 0, 1, 0);
    k_gru_step<<<CEILDIV(512000L, TPB), TPB, 0, stream>>>(gi, gh, bih, bhh, hbuf, R2, t);
  }
  k_bn_stats<<<128, 256, 0, stream>>>(R2, 96000, 128, meanb, rstdb, 1e-5f);
  k_bn_apply<<<CEILDIV(E12M, TPB), TPB, 0, stream>>>(R2, meanb, rstdb, bngg, bngb, E12M, 128, 1);

  // ================= attention: V=Rtmp, Q=R0, K=R1, xa->R2, xo->R0
  gemm(stream, R2, vw, Rtmp, 96000, 128, 128, 128, 128, 128, 0, 1, 0);
  k_conv_t<<<CEILDIV(96000L * 128, TPB), TPB, 0, stream>>>(R2, cqw, cqb, R0);
  k_conv_t<<<CEILDIV(96000L * 128, TPB), TPB, 0, stream>>>(R2, ckw, ckb, R1);
  k_attn<<<32000, 32, 0, stream>>>(R0, R1, Rtmp, vb, R2);
  gemm(stream, R2, ow, R0, 96000, 128, 128, 128, 128, 128, 0, 1, 0);
  k_fcreduce<<<CEILDIV(512000L, TPB), TPB, 0, stream>>>(R0, ob, fcw, fcb, out2f);

  // ================= residual 2
  gemm(stream, out2, r2w, R1, 96000, 128, 128, 128, 128, 128, 0, 1, 0);
  k_resreduce<<<CEILDIV(512000L, TPB), TPB, 0, stream>>>(R1, r2b, rcw, rcb, resred);
  k_bn_stats<<<128, 256, 0, stream>>>(resred, 4000, 128, meanb, rstdb, 1e-5f);
  k_bn_apply<<<CEILDIV(512000L, TPB), TPB, 0, stream>>>(resred, meanb, rstdb, bnrg, bnrb,
                                                        512000, 128, 0);

  // ================= time embedding + final projection + grid map
  k_te<<<4, 256, 0, stream>>>(ttime, t1w, t1b, t2w, t2b, te);
  k_prefin<<<CEILDIV(512000L, TPB), TPB, 0, stream>>>(out2f, resred, te, prefin);
  gemm(stream, prefin, finw, finout, 4000, 64, 128, 128, 128, 64, 0, 1, 0);
  k_bias<<<CEILDIV(256000L, TPB), TPB, 0, stream>>>(finout, finb, 256000, 64);
  gemm(stream, gmap, finout, gout, 500, 64, 500, 500, 64, 64,
       0, 0, 0, 8, 1, 0, 0, 500L * 64, 0, 500L * 64, 0);
  k_out<<<CEILDIV(256000L, TPB), TPB, 0, stream>>>(gout, out);

  (void)in_sizes; (void)n_in; (void)out_size;
}
